// GRU3D_78932908966246
// MI455X (gfx1250) — matrix operand order repaired and checked by reading, not compiled, not run on hardware
//
#include <hip/hip_runtime.h>
#include <hip/hip_bf16.h>
#include <math.h>

typedef __attribute__((ext_vector_type(16))) _Float16 v16h;
typedef __attribute__((ext_vector_type(8)))  float    v8f;

#define BATCH 2
#define NTOT  65536
#define KNN   4
#define HID   64
#define CIN   128
#define TILE  32   // points per block in gate kernels

// fast activations: v_exp_f32 + v_rcp_f32, saturate correctly at +/-inf
__device__ __forceinline__ float sigmoid_f(float v) {
  return __builtin_amdgcn_rcpf(1.0f + __expf(-v));
}
__device__ __forceinline__ float tanh_f(float v) {
  const float e = __expf(2.0f * v);
  return 1.0f - 2.0f * __builtin_amdgcn_rcpf(e + 1.0f);
}

// ---------------- Kernel A: pack/transpose h,x [C,N] -> hxT [b][n][c] ----------------
__global__ __launch_bounds__(256) void k_pack(const float* __restrict__ h,
                                              const float* __restrict__ x,
                                              float* __restrict__ hxT) {
  __shared__ float tile[32][33];
  const int b  = blockIdx.z;
  const int n0 = blockIdx.x * 32;
  const int c0 = blockIdx.y * 32;
  const int tx = threadIdx.x, ty = threadIdx.y;
  const float* src = (c0 < HID) ? (h + ((size_t)b * HID + c0) * NTOT)
                                : (x + ((size_t)b * HID + (c0 - HID)) * NTOT);
#pragma unroll
  for (int i = 0; i < 32; i += 8)
    tile[ty + i][tx] = src[(size_t)(ty + i) * NTOT + n0 + tx];
  __syncthreads();
#pragma unroll
  for (int i = 0; i < 32; i += 8) {
    const int n = n0 + ty + i;
    hxT[((size_t)b * NTOT + n) * CIN + c0 + tx] = tile[tx][ty + i];
  }
}

// ---------------- Kernel A2: pre-convert w_out (3 gates) to f16 A-fragment layout ----
// wA[((gate*4 + mt)*4 + kb)*32 + lane][e] , K=kb*32+hb*16+e
// where Mrow = mt*16 + (lane&15), hb = lane>>4.
__global__ __launch_bounds__(256) void k_cvtA(const float* __restrict__ wz_out,
                                              const float* __restrict__ wr_out,
                                              const float* __restrict__ wq_out,
                                              _Float16* __restrict__ wA) {
  const int t = blockIdx.x * 256 + threadIdx.x;   // 0 .. 1535
  const int gate = t >> 9;
  const int mt   = (t >> 7) & 3;
  const int kb   = (t >> 5) & 3;
  const int lane = t & 31;
  const int hb = lane >> 4, r16 = lane & 15;
  const float* wout = (gate == 0) ? wz_out : (gate == 1) ? wr_out : wq_out;
  const float* rp = wout + (mt * 16 + r16) * CIN + kb * 32 + hb * 16;
  v16h va;
#pragma unroll
  for (int e = 0; e < 8; ++e) {
    va[e]     = (_Float16)rp[e];
    va[8 + e] = (_Float16)rp[8 + e];
  }
  *(v16h*)(wA + (size_t)t * 16) = va;
}

// ---------------- Kernel B: z & r gates ----------------
__global__ __launch_bounds__(256) void k_zr(
    const float* __restrict__ xyz, const int* __restrict__ idx,
    const float* __restrict__ hxT, const float* __restrict__ h,
    const float* __restrict__ wz_pos, const float* __restrict__ bz_pos,
    const float* __restrict__ bz_out,
    const float* __restrict__ wr_pos, const float* __restrict__ br_pos,
    const float* __restrict__ br_out,
    const _Float16* __restrict__ wA,
    float* __restrict__ zbuf, float* __restrict__ rhT) {
  __shared__ float4 sW[2][CIN];                                 // {wx,wy,wz,bias} per channel
  __shared__ __align__(32) _Float16 sB[2][2][4][32][16];        // [gate][nt][kb][lane][e]

  const int tid = threadIdx.x;
  const int b   = blockIdx.y;
  const int n0  = blockIdx.x * TILE;

  // stage 0: positional weights -> LDS
  if (tid < 128) {
    const int c = tid;
    sW[0][c] = make_float4(wz_pos[3 * c], wz_pos[3 * c + 1], wz_pos[3 * c + 2], bz_pos[c]);
  } else {
    const int c = tid - 128;
    sW[1][c] = make_float4(wr_pos[3 * c], wr_pos[3 * c + 1], wr_pos[3 * c + 2], br_pos[c]);
  }
  __syncthreads();

  // stage 1: gather + depthwise reduce. 8 threads per point, 16 channels each.
  {
    const int nl = tid >> 3;   // point within tile
    const int ct = tid & 7;    // channel block
    const int c0 = ct * 16;
    const int n  = n0 + nl;
    const int4 nb = *(const int4*)(idx + ((size_t)b * NTOT + n) * KNN);
    const int j[4] = {nb.x, nb.y, nb.z, nb.w};
    const float cx = xyz[((size_t)b * 3 + 0) * NTOT + n];
    const float cy = xyz[((size_t)b * 3 + 1) * NTOT + n];
    const float cz = xyz[((size_t)b * 3 + 2) * NTOT + n];
    float rx[4], ry[4], rz[4];
    const float4* gp[4];
#pragma unroll
    for (int k = 0; k < 4; ++k) {
      rx[k] = xyz[((size_t)b * 3 + 0) * NTOT + j[k]] - cx;
      ry[k] = xyz[((size_t)b * 3 + 1) * NTOT + j[k]] - cy;
      rz[k] = xyz[((size_t)b * 3 + 2) * NTOT + j[k]] - cz;
      gp[k] = (const float4*)(hxT + ((size_t)b * NTOT + j[k]) * CIN + c0);
    }
    float accZ[16], accR[16];
#pragma unroll
    for (int e = 0; e < 16; ++e) { accZ[e] = 0.f; accR[e] = 0.f; }
#pragma unroll
    for (int q4 = 0; q4 < 4; ++q4) {
      float4 wz[4], wr[4];
#pragma unroll
      for (int e = 0; e < 4; ++e) {
        wz[e] = sW[0][c0 + q4 * 4 + e];
        wr[e] = sW[1][c0 + q4 * 4 + e];
      }
#pragma unroll
      for (int k = 0; k < 4; ++k) {
        const float4 g = gp[k][q4];
        const float gv[4] = {g.x, g.y, g.z, g.w};
#pragma unroll
        for (int e = 0; e < 4; ++e) {
          const float wzv = fmaf(wz[e].x, rx[k], fmaf(wz[e].y, ry[k], fmaf(wz[e].z, rz[k], wz[e].w)));
          const float wrv = fmaf(wr[e].x, rx[k], fmaf(wr[e].y, ry[k], fmaf(wr[e].z, rz[k], wr[e].w)));
          accZ[q4 * 4 + e] = fmaf(wzv, gv[e], accZ[q4 * 4 + e]);
          accR[q4 * 4 + e] = fmaf(wrv, gv[e], accR[q4 * 4 + e]);
        }
      }
    }
    // pack into WMMA B-layout f16 tiles: lane L holds K=(L/16)*16+e, col=L%16
    const int nt = nl >> 4, col = nl & 15;
    const int kb = ct >> 1, hb = ct & 1;
    const int lane = hb * 16 + col;
    v16h vz, vr;
#pragma unroll
    for (int e = 0; e < 16; ++e) { vz[e] = (_Float16)accZ[e]; vr[e] = (_Float16)accR[e]; }
    *(v16h*)&sB[0][nt][kb][lane][0] = vz;
    *(v16h*)&sB[1][nt][kb][lane][0] = vr;
  }
  __syncthreads();

  // stage 2: projection GEMM via v_wmma_f32_16x16x32_f16 + activation epilogue
  {
    const int wvid = tid >> 5;        // wave 0..7
    const int lane = tid & 31;
    const int gate = wvid >> 2;       // 0=z, 1=r
    const int mt   = wvid & 3;        // M tile (output channels)
    const int hb = lane >> 4, r16 = lane & 15;
    const float* bout = gate ? br_out : bz_out;
    // pre-converted A fragments: 2 x b128 per K-step
    const _Float16* aBase = wA + (size_t)(((gate * 4 + mt) * 4) * 32 + lane) * 16;
    v16h va[4];
#pragma unroll
    for (int kb = 0; kb < 4; ++kb)
      va[kb] = *(const v16h*)(aBase + (size_t)kb * 32 * 16);

    v8f bias;
#pragma unroll
    for (int i = 0; i < 8; ++i) bias[i] = bout[mt * 16 + hb * 8 + i];

#pragma unroll
    for (int nt = 0; nt < 2; ++nt) {
      v8f acc = bias;
#pragma unroll
      for (int kb = 0; kb < 4; ++kb) {
        const v16h vb = *(const v16h*)&sB[gate][nt][kb][lane][0];
        acc = __builtin_amdgcn_wmma_f32_16x16x32_f16(false, va[kb], false, vb,
                                                     (short)0, acc, false, false);
      }
      const int obase = mt * 16 + hb * 8;        // D layout: elem i -> M=obase+i, N=r16
      const int n = n0 + nt * 16 + r16;
      if (gate == 0) {
#pragma unroll
        for (int i = 0; i < 8; ++i)
          zbuf[((size_t)b * HID + obase + i) * NTOT + n] = sigmoid_f(acc[i]);
      } else {
        float o8[8];
#pragma unroll
        for (int i = 0; i < 8; ++i) {
          const float rv = sigmoid_f(acc[i]);
          const float hv = h[((size_t)b * HID + obase + i) * NTOT + n];
          o8[i] = rv * hv;
        }
        float* dst = rhT + ((size_t)b * NTOT + n) * HID + obase;
        *(float4*)(dst)     = make_float4(o8[0], o8[1], o8[2], o8[3]);
        *(float4*)(dst + 4) = make_float4(o8[4], o8[5], o8[6], o8[7]);
      }
    }
  }
}

// ---------------- Kernel C: q gate + final GRU combine ----------------
__global__ __launch_bounds__(256) void k_q(
    const float* __restrict__ xyz, const int* __restrict__ idx,
    const float* __restrict__ hxT, const float* __restrict__ rhT,
    const float* __restrict__ h,
    const float* __restrict__ wq_pos, const float* __restrict__ bq_pos,
    const float* __restrict__ bq_out,
    const _Float16* __restrict__ wA,
    const float* __restrict__ zbuf, float* __restrict__ out) {
  __shared__ float4 sW[CIN];
  __shared__ __align__(32) _Float16 sB[2][4][32][16];           // [nt][kb][lane][e]

  const int tid = threadIdx.x;
  const int b   = blockIdx.y;
  const int n0  = blockIdx.x * TILE;

  if (tid < 128)
    sW[tid] = make_float4(wq_pos[3 * tid], wq_pos[3 * tid + 1], wq_pos[3 * tid + 2], bq_pos[tid]);
  __syncthreads();

  {
    const int nl = tid >> 3;
    const int ct = tid & 7;
    const int c0 = ct * 16;     // c0<64 -> rhT (r*h) ; c0>=64 -> x half of hxT
    const int n  = n0 + nl;
    const int4 nb = *(const int4*)(idx + ((size_t)b * NTOT + n) * KNN);
    const int j[4] = {nb.x, nb.y, nb.z, nb.w};
    const float cx = xyz[((size_t)b * 3 + 0) * NTOT + n];
    const float cy = xyz[((size_t)b * 3 + 1) * NTOT + n];
    const float cz = xyz[((size_t)b * 3 + 2) * NTOT + n];
    float rx[4], ry[4], rz[4];
    const float4* gp[4];
#pragma unroll
    for (int k = 0; k < 4; ++k) {
      rx[k] = xyz[((size_t)b * 3 + 0) * NTOT + j[k]] - cx;
      ry[k] = xyz[((size_t)b * 3 + 1) * NTOT + j[k]] - cy;
      rz[k] = xyz[((size_t)b * 3 + 2) * NTOT + j[k]] - cz;
      const float* base = (ct < 4)
          ? (rhT + ((size_t)b * NTOT + j[k]) * HID + c0)
          : (hxT + ((size_t)b * NTOT + j[k]) * CIN + c0);
      gp[k] = (const float4*)base;
    }
    float accQ[16];
#pragma unroll
    for (int e = 0; e < 16; ++e) accQ[e] = 0.f;
#pragma unroll
    for (int q4 = 0; q4 < 4; ++q4) {
      float4 wq[4];
#pragma unroll
      for (int e = 0; e < 4; ++e) wq[e] = sW[c0 + q4 * 4 + e];
#pragma unroll
      for (int k = 0; k < 4; ++k) {
        const float4 g = gp[k][q4];
        const float gv[4] = {g.x, g.y, g.z, g.w};
#pragma unroll
        for (int e = 0; e < 4; ++e) {
          const float wvq = fmaf(wq[e].x, rx[k], fmaf(wq[e].y, ry[k], fmaf(wq[e].z, rz[k], wq[e].w)));
          accQ[q4 * 4 + e] = fmaf(wvq, gv[e], accQ[q4 * 4 + e]);
        }
      }
    }
    const int nt = nl >> 4, col = nl & 15;
    const int kb = ct >> 1, hb = ct & 1;
    const int lane = hb * 16 + col;
    v16h vq;
#pragma unroll
    for (int e = 0; e < 16; ++e) vq[e] = (_Float16)accQ[e];
    *(v16h*)&sB[nt][kb][lane][0] = vq;
  }
  __syncthreads();

  {
    const int wvid = tid >> 5;
    const int lane = tid & 31;
    const int mt = wvid & 3;
    const int ntile = wvid >> 2;
    const int hb = lane >> 4, r16 = lane & 15;
    const _Float16* aBase = wA + (size_t)(((2 * 4 + mt) * 4) * 32 + lane) * 16;  // gate q = 2
    v16h va[4];
#pragma unroll
    for (int kb = 0; kb < 4; ++kb)
      va[kb] = *(const v16h*)(aBase + (size_t)kb * 32 * 16);
    v8f acc;
#pragma unroll
    for (int i = 0; i < 8; ++i) acc[i] = bq_out[mt * 16 + hb * 8 + i];
#pragma unroll
    for (int kb = 0; kb < 4; ++kb) {
      const v16h vb = *(const v16h*)&sB[ntile][kb][lane][0];
      acc = __builtin_amdgcn_wmma_f32_16x16x32_f16(false, va[kb], false, vb,
                                                   (short)0, acc, false, false);
    }
    const int obase = mt * 16 + hb * 8;
    const int n = n0 + ntile * 16 + r16;
#pragma unroll
    for (int i = 0; i < 8; ++i) {
      const size_t off = ((size_t)b * HID + obase + i) * NTOT + n;
      const float q  = tanh_f(acc[i]);
      const float zv = zbuf[off];
      const float hv = h[off];
      out[off] = (1.0f - zv) * hv + zv * q;
    }
  }
}

extern "C" void kernel_launch(void* const* d_in, const int* in_sizes, int n_in,
                              void* d_out, int out_size, void* d_ws, size_t ws_size,
                              hipStream_t stream) {
  (void)in_sizes; (void)n_in; (void)out_size; (void)ws_size;
  const float* xyz    = (const float*)d_in[0];
  const float* h      = (const float*)d_in[1];
  const float* x      = (const float*)d_in[2];
  const int*   knn    = (const int*)d_in[3];
  const float* wz_pos = (const float*)d_in[4];
  const float* bz_pos = (const float*)d_in[5];
  const float* wz_out = (const float*)d_in[6];
  const float* bz_out = (const float*)d_in[7];
  const float* wr_pos = (const float*)d_in[8];
  const float* br_pos = (const float*)d_in[9];
  const float* wr_out = (const float*)d_in[10];
  const float* br_out = (const float*)d_in[11];
  const float* wq_pos = (const float*)d_in[12];
  const float* bq_pos = (const float*)d_in[13];
  const float* wq_out = (const float*)d_in[14];
  const float* bq_out = (const float*)d_in[15];
  float* out = (float*)d_out;

  // workspace: hxT [B,N,128] (64MB) | rhT [B,N,64] (32MB) | zbuf [B,64,N] (32MB)
  //            | wA f16 A-fragments 3*4*4*32*16 halfs (48KB)
  float* hxT  = (float*)d_ws;
  float* rhT  = hxT + (size_t)BATCH * NTOT * CIN;
  float* zbuf = rhT + (size_t)BATCH * NTOT * HID;
  _Float16* wA = (_Float16*)(zbuf + (size_t)BATCH * HID * NTOT);

  dim3 gA(NTOT / 32, CIN / 32, BATCH);
  dim3 bA(32, 8);
  k_pack<<<gA, bA, 0, stream>>>(h, x, hxT);
  k_cvtA<<<6, 256, 0, stream>>>(wz_out, wr_out, wq_out, wA);

  dim3 gG(NTOT / TILE, BATCH);
  k_zr<<<gG, 256, 0, stream>>>(xyz, knn, hxT, h,
                               wz_pos, bz_pos, bz_out,
                               wr_pos, br_pos, br_out,
                               wA, zbuf, rhT);
  k_q<<<gG, 256, 0, stream>>>(xyz, knn, hxT, rhT, h,
                              wq_pos, bq_pos, bq_out,
                              wA, zbuf, out);
}
